// CBNNConv2d_51410758533594
// MI455X (gfx1250) — 8-bit matrix operand order repaired and checked by reading, compiled, not run on hardware
//
#include <hip/hip_runtime.h>
#include <hip/hip_bf16.h>

typedef __attribute__((ext_vector_type(16))) int   v16i;
typedef __attribute__((ext_vector_type(4)))  int   v4i;
typedef __attribute__((ext_vector_type(4)))  float v4f;
typedef __attribute__((ext_vector_type(8)))  float v8f;

namespace {
constexpr int NB   = 32;
constexpr int CIN  = 256;
constexpr int COUT = 256;
constexpr int HH   = 56;
constexpr int WW   = 56;
constexpr int PH   = 58;
constexpr int PW   = 58;
constexpr int PIX  = HH * WW;
constexpr int PPIX = PH * PW;
constexpr int NPOS = NB * PIX;
constexpr int KCHUNKS = 9 * (CIN / 128);
constexpr int KBITS = 12;

constexpr unsigned char FP8_P1 = 0x38;
constexpr unsigned char FP8_M1 = 0xB8;

constexpr size_t XQP_BYTES = (size_t)NB * PPIX * CIN;
}

__global__ void zero_xq(v4i* __restrict__ p) {
  p[(size_t)blockIdx.x * 256 + threadIdx.x] = (v4i){0, 0, 0, 0};
}

__global__ __launch_bounds__(256)
void binarize_x(const float* __restrict__ x, unsigned char* __restrict__ xq) {
  __shared__ __align__(16) unsigned char t[64 * 64];

  const int b   = blockIdx.x;
  const int pb  = b % 49;
  const int cbk = (b / 49) & 3;
  const int n   = b / (49 * 4);
  const int p0  = pb * 64;
  const int ci0 = cbk * 64;

  const int r   = threadIdx.x >> 2;
  const int seg = threadIdx.x & 3;

  const v4f* src = (const v4f*)(x + ((size_t)(n * CIN + ci0 + r)) * PIX + p0 + seg * 16);
#pragma unroll
  for (int g = 0; g < 4; ++g) {
    v4f f = src[g];
#pragma unroll
    for (int j = 0; j < 4; ++j) {
      float v = f[j];
      unsigned char sb = (v > 0.f) ? FP8_P1 : ((v < 0.f) ? FP8_M1 : (unsigned char)0);
      t[(seg * 16 + g * 4 + j) * 64 + r] = sb;
    }
  }
  __syncthreads();

  const int q  = threadIdx.x >> 2;
  const int p  = p0 + q;
  const int py = p / WW;
  const int px = p - py * WW;
  v4i d = *(const v4i*)&t[q * 64 + seg * 16];
  *(v4i*)(xq + (((size_t)n * PH + (py + 1)) * PW + (px + 1)) * CIN + ci0 + seg * 16) = d;
}

__global__ void decode_w(const float* __restrict__ cb, const int* __restrict__ codes,
                         unsigned char* __restrict__ wp) {
  int i = blockIdx.x * 256 + threadIdx.x;
  int oc = i / (CIN * 9);
  int r  = i - oc * (CIN * 9);
  int ci = r / 9;
  int k9 = r - ci * 9;
  int codeIdx = i / KBITS;
  int bit     = i - codeIdx * KBITS;
  float v = cb[codes[codeIdx] * KBITS + bit];
  unsigned char b = (v >= 0.f) ? FP8_P1 : FP8_M1;

  int chunk = k9 * 2 + (ci >> 7);
  int kloc  = ci & 127;
  int blk8  = kloc >> 4;
  int par   = blk8 & 1;
  size_t dst = (((size_t)chunk * COUT + oc) * 2 + par) * 64
             + (size_t)(blk8 >> 1) * 16 + (kloc & 15);
  wp[dst] = b;
}

__device__ __forceinline__ void stage32_async(const unsigned char* src,
                                              const unsigned char* ldsp) {
  const unsigned ldso = (unsigned)(uintptr_t)ldsp;
  asm volatile("global_load_async_to_lds_b128 %0, %1, off"
               :: "v"(ldso), "v"(src), "v"(ldsp) : "memory");
  asm volatile("global_load_async_to_lds_b128 %0, %1, off offset:16"
               :: "v"(ldso), "v"(src), "v"(ldsp) : "memory");
}

__global__ __launch_bounds__(256, 2)
void bconv_fp8_wmma(const unsigned char* __restrict__ xq,
                    const unsigned char* __restrict__ wp,
                    float* __restrict__ out) {
  __shared__ __align__(16) unsigned char ldsB[2 * 64 * 128];

  const int tid  = threadIdx.x;
  const int wave = tid >> 5;
  const int lane = tid & 31;

  const int mblk = blockIdx.x & 1;
  const int nblk = blockIdx.x >> 1;
  const int oc_base = mblk * 128 + wave * 16;
  const int p0 = nblk * 64;

  const int spos = tid >> 2;
  const int sq   = tid & 3;
  const int sp   = p0 + spos;
  const int sn   = sp / PIX;
  const int srr  = sp - sn * PIX;
  const int syy  = srr / WW;
  const int sxx  = srr - syy * WW;
  const unsigned char* tbase =
      xq + (((size_t)sn * PPIX + (size_t)syy * PW + sxx) * CIN) + sq * 32;
  unsigned char* lbase = ldsB + spos * 128 + sq * 32;

  const int bn = lane & 15;
  const int bh = lane >> 4;

  v8f acc0 = {}, acc1 = {}, acc2 = {}, acc3 = {};

  auto issue = [&](int c) {
    const int kpos = c >> 1;
    const int half = c & 1;
    const int kh = kpos / 3;
    const int kw = kpos - kh * 3;
    const unsigned goff = (unsigned)((kh * PW + kw) * CIN + half * 128);
    stage32_async(tbase + goff, lbase + (c & 1) * 8192);
  };

  issue(0);

  for (int c = 0; c < KCHUNKS; ++c) {
    if (c + 1 < KCHUNKS) {
      issue(c + 1);
      asm volatile("s_wait_asynccnt 0x2" ::: "memory");
    } else {
      asm volatile("s_wait_asynccnt 0x0" ::: "memory");
    }
    __syncthreads();

    const unsigned char* lb = ldsB + (c & 1) * 8192;

    union { v4i q[4]; v16i v; } af;
    const v4i* ap = (const v4i*)(wp + (((size_t)c * COUT + oc_base + (lane & 15)) * 2
                                       + (lane >> 4)) * 64);
    af.q[0] = ap[0]; af.q[1] = ap[1]; af.q[2] = ap[2]; af.q[3] = ap[3];
    if (c + 1 < KCHUNKS) {
      __builtin_prefetch(ap + 32 * COUT / 4, 0, 1);
    }

    union { v4i q[4]; v16i v; } bf;
#pragma unroll
    for (int sub = 0; sub < 4; ++sub) {
      const unsigned char* bb = lb + (sub * 16 + bn) * 128 + bh * 16;
      bf.q[0] = *(const v4i*)(bb + 0);
      bf.q[1] = *(const v4i*)(bb + 32);
      bf.q[2] = *(const v4i*)(bb + 64);
      bf.q[3] = *(const v4i*)(bb + 96);
      v8f& acc = (sub == 0) ? acc0 : (sub == 1) ? acc1 : (sub == 2) ? acc2 : acc3;
      acc = __builtin_amdgcn_wmma_f32_16x16x128_fp8_fp8(af.v, bf.v, (short)0, acc,
                                                        false, false);
    }
    __syncthreads();
  }

  const int mrow0 = oc_base + (lane >> 4) * 8;
#pragma unroll
  for (int sub = 0; sub < 4; ++sub) {
    const v8f& acc = (sub == 0) ? acc0 : (sub == 1) ? acc1 : (sub == 2) ? acc2 : acc3;
    const int p  = p0 + sub * 16 + bn;
    const int n  = p / PIX;
    const int rr = p - n * PIX;
    float* o = out + ((size_t)n * COUT + mrow0) * PIX + rr;
#pragma unroll
    for (int v = 0; v < 8; ++v) o[(size_t)v * PIX] = acc[v];
  }
}

extern "C" void kernel_launch(void* const* d_in, const int* in_sizes, int n_in,
                              void* d_out, int out_size, void* d_ws, size_t ws_size,
                              hipStream_t stream) {
  const float* x   = (const float*)d_in[0];
  const float* wlt = (const float*)d_in[1];
  const float* cb  = (const float*)d_in[2];
  const int*   ev  = (const int*)d_in[3];
  (void)wlt; (void)in_sizes; (void)n_in; (void)out_size; (void)ws_size;

  unsigned char* xq = (unsigned char*)d_ws;
  unsigned char* wp = (unsigned char*)d_ws + XQP_BYTES;

  zero_xq<<<(int)(XQP_BYTES / 16 / 256), 256, 0, stream>>>((v4i*)xq);
  binarize_x<<<NB * 4 * 49, 256, 0, stream>>>(x, xq);
  decode_w<<<(COUT * CIN * 9) / 256, 256, 0, stream>>>(cb, ev, wp);

  bconv_fp8_wmma<<<2 * (NPOS / 64), 256, 0, stream>>>(xq, wp, (float*)d_out);
}
